// REDFM_46548855554443
// MI455X (gfx1250) — matrix operand order repaired and checked by reading, not compiled, not run on hardware
//
#include <hip/hip_runtime.h>
#include <hip/hip_bf16.h>

typedef __attribute__((ext_vector_type(16))) _Float16 v16h;
typedef __attribute__((ext_vector_type(8)))  float    v8f;

#define NPIX 16384
#define NCH  256
#define WIDTH 128
#define HEIGHT 128
#define NTILES (NPIX / 16)   // 1024 16-row/col tiles (even -> clean unroll-by-2)
#define KTILES (NCH / 32)    // 8 K-tiles of 32
#define TILE_ELEMS (KTILES * 32)   // 256 v16h elements = 8KB per column tile

// ---------------- Stage 1: per-pixel inverse L2 norm over channels ----------
__global__ __launch_bounds__(256) void REDFM_norm_kernel(const float* __restrict__ x,
                                                         float* __restrict__ invn) {
    int p = blockIdx.x * blockDim.x + threadIdx.x;
    if (p >= NPIX) return;
    float s = 0.f;
    #pragma unroll 8
    for (int c = 0; c < NCH; ++c) {
        float v = x[c * NPIX + p];   // fixed c: contiguous in p -> coalesced
        s += v * v;
    }
    invn[p] = rsqrtf(s);
}

// ---------------- Stage 2: normalize + f16 + fragment-ready packing ---------
// role 0: WMMA A-matrix 16x32 f16 layout (lane<16: K {0..7,16..23}; lane>=16: K {8..15,24..31})
// role 1: WMMA B-matrix 32x16 f16 layout (lane<16: K {0..7,16..23}; lane>=16: K {8..15,24..31})
// Packed element for (tile t, ktile kt, lane L) is 16 contiguous halves:
//   out[((t*KTILES + kt)*32 + L)*16 + h]
__global__ __launch_bounds__(256) void REDFM_pack_kernel(const float* __restrict__ x,
                                                         const float* __restrict__ invn,
                                                         _Float16* __restrict__ out,
                                                         int role) {
    int tid  = blockIdx.x * blockDim.x + threadIdx.x;   // NTILES*KTILES*32 threads
    int lane = tid & 31;
    int kt   = (tid >> 5) & (KTILES - 1);
    int t    = tid >> 8;
    if (t >= NTILES) return;

    int k0 = kt * 32;
    int p  = t * 16 + (lane & 15);       // pixel (row for A-role, col for B-role)
    float inv = invn[p];
    int hi = (lane >= 16) ? 1 : 0;

    v16h vals;
    #pragma unroll
    for (int h = 0; h < 16; ++h) {
        int k;
        if (role == 0) {   // A-role K map
            k = (h < 8) ? (8 * hi + h) : (16 + 8 * hi + (h - 8));
        } else {           // B-role K map
            k = (h < 8) ? (8 * hi + h) : (16 + 8 * hi + (h - 8));
        }
        vals[h] = (_Float16)(x[(k0 + k) * NPIX + p] * inv);
    }
    *(v16h*)(out + (size_t)tid * 16) = vals;
}

// ---------------- Stage 3: fused GEMM + running top-2 per row ---------------
__device__ __forceinline__ void REDFM_tile_step(const v16h* lbuf, int lane, int colBase,
                                                const v16h* a,
                                                float* b1, float* b2, int* ix) {
    // Pull all 8 B fragments from LDS up front (batched ds_load_b128s)
    v16h b[KTILES];
    #pragma unroll
    for (int kt = 0; kt < KTILES; ++kt)
        b[kt] = lbuf[kt * 32 + lane];

    // Two independent accumulation chains for WMMA ILP
    v8f acc0 = {}, acc1 = {};
    #pragma unroll
    for (int kt = 0; kt < KTILES / 2; ++kt) {
        acc0 = __builtin_amdgcn_wmma_f32_16x16x32_f16(
                   false, a[kt],              false, b[kt],
                   (short)0, acc0, false, false);
        acc1 = __builtin_amdgcn_wmma_f32_16x16x32_f16(
                   false, a[kt + KTILES / 2], false, b[kt + KTILES / 2],
                   (short)0, acc1, false, false);
    }

    // C/D layout: VGPR r, lane -> (row = r + 8*(lane>=16), col = colBase + lane%16)
    int col = colBase + (lane & 15);
    #pragma unroll
    for (int r = 0; r < 8; ++r) {
        float v   = acc0[r] + acc1[r];
        bool  gt  = v > b1[r];
        float old = b1[r];
        b1[r] = gt ? v   : b1[r];
        ix[r] = gt ? col : ix[r];
        float cand = gt ? old : v;
        b2[r] = cand > b2[r] ? cand : b2[r];
    }
}

// One wave per 16-row strip; 8 waves/block; 128 blocks cover all 1024 strips.
// A fragments (16x256) register-resident. The whole block shares one column
// sweep of B via double-buffered LDS staging (2 x 8KB): each block reads each
// B tile from L2 exactly once. Column loop unrolled by 2 with two distinct
// staging register sets so the next tile's global load stays in flight across
// a full tile of WMMA compute (no loadcnt wait at the barrier).
__global__ __launch_bounds__(256, 1) void REDFM_simtop2_kernel(
        const v16h* __restrict__ Ap,   // A-role packing of the "row" matrix
        const v16h* __restrict__ Bp,   // B-role packing of the "col" matrix
        float* __restrict__ v1, int* __restrict__ i1, float* __restrict__ v2) {
    __shared__ v16h ldsb[2][TILE_ELEMS];

    int tid  = threadIdx.x;
    int lane = tid & 31;
    int wave = tid >> 5;
    int it   = blockIdx.x * 8 + wave;          // row tile 0..1023

    // Register-resident A fragments for the whole K=256
    v16h a[KTILES];
    #pragma unroll
    for (int kt = 0; kt < KTILES; ++kt)
        a[kt] = Ap[(size_t)(it * KTILES + kt) * 32 + lane];

    float b1[8], b2[8];
    int   ix[8];
    #pragma unroll
    for (int r = 0; r < 8; ++r) { b1[r] = -1e30f; b2[r] = -1e30f; ix[r] = 0; }

    // Prologue: fetch tile 0 (one 32B element per thread)
    v16h stage0 = Bp[tid];
    v16h stage1;

    for (int jt = 0; jt < NTILES; jt += 2) {
        // ---- even tile: commit stage0, prefetch jt+1 into stage1 ----
        ldsb[0][tid] = stage0;
        stage1 = Bp[(size_t)(jt + 1) * TILE_ELEMS + tid];   // NTILES even: always valid
        __syncthreads();
        REDFM_tile_step(&ldsb[0][0], lane, jt * 16, a, b1, b2, ix);

        // ---- odd tile: commit stage1, prefetch jt+2 into stage0 ----
        ldsb[1][tid] = stage1;
        if (jt + 2 < NTILES)
            stage0 = Bp[(size_t)(jt + 2) * TILE_ELEMS + tid];
        __syncthreads();
        REDFM_tile_step(&ldsb[1][0], lane, (jt + 1) * 16, a, b1, b2, ix);
    }

    // Cross-lane top-2 merge within each 16-lane half (wave32)
    #pragma unroll
    for (int r = 0; r < 8; ++r) {
        float B1 = b1[r], B2 = b2[r];
        int   I1 = ix[r];
        #pragma unroll
        for (int off = 8; off >= 1; off >>= 1) {
            float oB1 = __shfl_xor(B1, off, 16);
            float oB2 = __shfl_xor(B2, off, 16);
            int   oI1 = __shfl_xor(I1, off, 16);
            if (oB1 > B1) { B2 = fmaxf(B1, oB2); B1 = oB1; I1 = oI1; }
            else          { B2 = fmaxf(B2, oB1); }
        }
        if ((lane & 15) == 0) {
            int row = it * 16 + r + ((lane >> 4) << 3);
            v1[row] = B1; i1[row] = I1; v2[row] = B2;
        }
    }
}

// ---------------- Stage 4: mutual-NN + ratio test + border ------------------
__global__ __launch_bounds__(256) void REDFM_finalize_kernel(
        const float* __restrict__ v1_12, const int* __restrict__ i1_12,
        const float* __restrict__ v2_12,
        const float* __restrict__ v1_21, const int* __restrict__ i1_21,
        const float* __restrict__ v2_21,
        float* __restrict__ out) {
    int i = blockIdx.x * blockDim.x + threadIdx.x;
    if (i >= NPIX) return;

    float d1 = sqrtf(fmaxf(2.f - 2.f * v1_12[i], 1e-12f));
    float d2 = sqrtf(fmaxf(2.f - 2.f * v2_12[i], 1e-12f));
    int   nn = i1_12[i];
    float r12 = d1 / (d2 + 1e-8f);

    bool  mutual = (i1_21[nn] == i);
    float e1 = sqrtf(fmaxf(2.f - 2.f * v1_21[nn], 1e-12f));
    float e2 = sqrtf(fmaxf(2.f - 2.f * v2_21[nn], 1e-12f));
    float r21 = e1 / (e2 + 1e-8f);

    float ratio = fmaxf(r12, r21);
    int xA = i  % WIDTH, yA = i  / WIDTH;
    int xB = nn % WIDTH, yB = nn / WIDTH;
    bool border = (xA == 0) | (xA == WIDTH - 1) | (yA == 0) | (yA == HEIGHT - 1) |
                  (xB == 0) | (xB == WIDTH - 1) | (yB == 0) | (yB == HEIGHT - 1);
    bool valid = mutual && (ratio < 0.95f) && !border;

    out[i * 2 + 0]   = valid ? d1 : 0.f;   // dist_out
    out[i * 2 + 1]   = valid ? d2 : 0.f;
    out[2 * NPIX + i] = (float)nn;         // nn12
    out[3 * NPIX + i] = valid ? 1.f : 0.f; // valid
}

extern "C" void kernel_launch(void* const* d_in, const int* in_sizes, int n_in,
                              void* d_out, int out_size, void* d_ws, size_t ws_size,
                              hipStream_t stream) {
    const float* A = (const float*)d_in[0];
    const float* B = (const float*)d_in[1];
    float* out = (float*)d_out;

    char* ws = (char*)d_ws;
    size_t off = 0;
    float*    invnA   = (float*)(ws + off); off += NPIX * sizeof(float);
    float*    invnB   = (float*)(ws + off); off += NPIX * sizeof(float);
    const size_t packBytes = (size_t)NPIX * NCH * sizeof(_Float16);   // 8 MB each
    _Float16* ApackA  = (_Float16*)(ws + off); off += packBytes;      // A, A-role
    _Float16* BpackB  = (_Float16*)(ws + off); off += packBytes;      // B, B-role
    _Float16* BpackA  = (_Float16*)(ws + off); off += packBytes;      // B, A-role
    _Float16* ApackB  = (_Float16*)(ws + off); off += packBytes;      // A, B-role
    float* v1_12 = (float*)(ws + off); off += NPIX * sizeof(float);
    int*   i1_12 = (int*)  (ws + off); off += NPIX * sizeof(int);
    float* v2_12 = (float*)(ws + off); off += NPIX * sizeof(float);
    float* v1_21 = (float*)(ws + off); off += NPIX * sizeof(float);
    int*   i1_21 = (int*)  (ws + off); off += NPIX * sizeof(int);
    float* v2_21 = (float*)(ws + off); off += NPIX * sizeof(float);

    dim3 blk(256);

    // Stage 1: norms
    REDFM_norm_kernel<<<NPIX / 256, blk, 0, stream>>>(A, invnA);
    REDFM_norm_kernel<<<NPIX / 256, blk, 0, stream>>>(B, invnB);

    // Stage 2: packing (NTILES*KTILES*32 = 262144 threads each)
    int packBlocks = (NTILES * KTILES * 32) / 256;
    REDFM_pack_kernel<<<packBlocks, blk, 0, stream>>>(A, invnA, ApackA, 0);
    REDFM_pack_kernel<<<packBlocks, blk, 0, stream>>>(B, invnB, BpackB, 1);
    REDFM_pack_kernel<<<packBlocks, blk, 0, stream>>>(B, invnB, BpackA, 0);
    REDFM_pack_kernel<<<packBlocks, blk, 0, stream>>>(A, invnA, ApackB, 1);

    // Stage 3: fused GEMM + top-2, both directions (1024 strips / 8 waves = 128 blocks)
    REDFM_simtop2_kernel<<<NTILES / 8, blk, 0, stream>>>(
        (const v16h*)ApackA, (const v16h*)BpackB, v1_12, i1_12, v2_12);
    REDFM_simtop2_kernel<<<NTILES / 8, blk, 0, stream>>>(
        (const v16h*)BpackA, (const v16h*)ApackB, v1_21, i1_21, v2_21);

    // Stage 4: finalize
    REDFM_finalize_kernel<<<NPIX / 256, blk, 0, stream>>>(
        v1_12, i1_12, v2_12, v1_21, i1_21, v2_21, out);
}
